// CausalSelfAttention_2654289789109
// MI455X (gfx1250) — matrix operand order repaired and checked by reading, compiled, not run on hardware
//
#include <hip/hip_runtime.h>

typedef __attribute__((ext_vector_type(16))) __bf16 v16bf;
typedef __attribute__((ext_vector_type(8)))  __bf16 v8bf;
typedef __attribute__((ext_vector_type(4)))  __bf16 v4bf;
typedef __attribute__((ext_vector_type(8)))  float  v8f;

#define DEVFN static __device__ __forceinline__

namespace cfg {
constexpr int B  = 2;
constexpr int T  = 2048;
constexpr int C  = 2048;     // d_model
constexpr int H  = 16;
constexpr int HK = 4;
constexpr int D  = 128;      // head_dim
constexpr int NQKV = C + 2 * HK * D;   // 3072 = 2048 q + 512 k + 512 v
constexpr float SCALE = 0.08838834764831845f;  // 1/sqrt(128)
}

// ---------------------------------------------------------------- WMMA core
DEVFN v8f wmma_bf16(v16bf a, v16bf b, v8f c) {
  // D = A(16x32 bf16) * B(32x16 bf16) + C(16x16 f32)
  return __builtin_amdgcn_wmma_f32_16x16x32_bf16(false, a, false, b,
                                                 (short)0, c, false, false);
}

// A-fragment (16x32 bf16): lane L holds row M=L&15; 16 contiguous K elems
// {hi*16..hi*16+15} where hi = L>>4.  rowk0 points at K=0 of this row.
DEVFN v16bf make_a_frag(const __bf16* rowk0, int hi) {
  return *(const v16bf*)(rowk0 + hi * 16);
}
// B-fragment (32x16 bf16): lane L holds column N=L&15, 16 contiguous K values
// starting at K = (L>>4)*16  -> a single 32-byte contiguous load per lane.

// ---------------------------------------------------------------- fp32->bf16
__global__ void cvt_f32_bf16(const float* __restrict__ in,
                             __bf16* __restrict__ out, int n4) {
  int i = blockIdx.x * blockDim.x + threadIdx.x;
  if (i >= n4) return;
  float4 v = ((const float4*)in)[i];
  v4bf o;
  o[0] = (__bf16)v.x; o[1] = (__bf16)v.y; o[2] = (__bf16)v.z; o[3] = (__bf16)v.w;
  ((v4bf*)out)[i] = o;
}

// ---------------------------------------------------------------- QKV GEMM
// out = x @ [Wq;Wkv].T ; per-wave 32x64 tile (2 A-frags share 4 B-frags ->
// 8 WMMAs per 12 b128 loads). Scatter into Qh[b,h,t,d], Kh[b,hk,t,d],
// Vt[b,hk,d,t] (V transposed so the P@V B-fragments are contiguous).
__global__ __launch_bounds__(128) void gemm_qkv(const __bf16* __restrict__ Xb,
                                                const __bf16* __restrict__ Wqkv,
                                                __bf16* __restrict__ Qh,
                                                __bf16* __restrict__ Kh,
                                                __bf16* __restrict__ Vt) {
  using namespace cfg;
  const int lane = threadIdx.x & 31;
  const int wave = threadIdx.x >> 5;
  const int hi = lane >> 4, r = lane & 15;
  const int mtiles = (B * T) / 32;                  // 128
  const int mtile = blockIdx.x % mtiles;
  const int nblk  = blockIdx.x / mtiles;            // 0..11
  const int m0 = mtile * 32;
  const int n0 = nblk * 256 + wave * 64;

  v8f acc[2][4];
#pragma unroll
  for (int mt = 0; mt < 2; ++mt)
#pragma unroll
    for (int nt = 0; nt < 4; ++nt)
#pragma unroll
      for (int i = 0; i < 8; ++i) acc[mt][nt][i] = 0.f;

  const __bf16* arow0 = Xb + (size_t)(m0 + r) * C;
  const __bf16* arow1 = Xb + (size_t)(m0 + 16 + r) * C;
  for (int k0 = 0; k0 < C; k0 += 32) {
    v16bf a0 = make_a_frag(arow0 + k0, hi);
    v16bf a1 = make_a_frag(arow1 + k0, hi);
#pragma unroll
    for (int nt = 0; nt < 4; ++nt) {
      const __bf16* brow = Wqkv + (size_t)(n0 + nt * 16 + r) * C + k0 + hi * 16;
      v16bf bfr = *(const v16bf*)brow;
      acc[0][nt] = wmma_bf16(a0, bfr, acc[0][nt]);
      acc[1][nt] = wmma_bf16(a1, bfr, acc[1][nt]);
    }
  }

#pragma unroll
  for (int mt = 0; mt < 2; ++mt)
#pragma unroll
    for (int nt = 0; nt < 4; ++nt) {
      const int n = n0 + nt * 16 + r;
#pragma unroll
      for (int rr = 0; rr < 8; ++rr) {
        const int m = m0 + mt * 16 + rr + 8 * hi;  // C-frag: VGPR rr -> row rr+8*hi
        const int b = m >> 11, t = m & (T - 1);
        const __bf16 val = (__bf16)acc[mt][nt][rr];
        if (n < C) {                              // Q
          const int h = n >> 7, d = n & 127;
          Qh[(((size_t)b * H + h) * T + t) * D + d] = val;
        } else if (n < C + HK * D) {              // K
          const int nn = n - C, hk = nn >> 7, d = nn & 127;
          Kh[(((size_t)b * HK + hk) * T + t) * D + d] = val;
        } else {                                  // V -> transposed [d][t]
          const int nn = n - C - HK * D, hk = nn >> 7, d = nn & 127;
          Vt[(((size_t)b * HK + hk) * D + d) * (size_t)T + t] = val;
        }
      }
    }
}

// ---------------------------------------------------------------- RoPE
__global__ void rope_inplace(__bf16* __restrict__ Qh, __bf16* __restrict__ Kh) {
  using namespace cfg;
  const int i = blockIdx.x * blockDim.x + threadIdx.x;
  const int nq = B * H * T * (D / 2);
  const int nk = B * HK * T * (D / 2);
  __bf16* base; int idx;
  if (i < nq)            { base = Qh; idx = i; }
  else if (i < nq + nk)  { base = Kh; idx = i - nq; }
  else return;
  const int d   = idx & 63;
  const int row = idx >> 6;                // (b*nh + h)*T + t
  const int t   = row & (T - 1);
  __bf16* p = base + (size_t)row * D;
  const float x1 = (float)p[d], x2 = (float)p[d + 64];
  const float inv_freq = __powf(10000.f, -(float)d * (1.f / 64.f));
  const float ang = (float)t * inv_freq;
  const float s = __sinf(ang), c = __cosf(ang);
  p[d]      = (__bf16)(x1 * c - x2 * s);
  p[d + 64] = (__bf16)(x2 * c + x1 * s);
}

// ---------------------------------------------------------------- Flash attn
// One wave per 16-row q tile; online softmax; P staged via LDS to re-shape
// C-fragment -> A-fragment for the P@V WMMAs.
__global__ __launch_bounds__(256) void attn_fa(const __bf16* __restrict__ Qh,
                                               const __bf16* __restrict__ Kh,
                                               const __bf16* __restrict__ Vt,
                                               __bf16* __restrict__ Y) {
  using namespace cfg;
  __shared__ __bf16 Pst[8][16][32];        // per-wave 16x32 P tile
  const int lane = threadIdx.x & 31;
  const int wave = threadIdx.x >> 5;
  const int hi = lane >> 4, r = lane & 15;
  const int w  = blockIdx.x * 8 + wave;    // 0..4095
  const int qt = w & (T / 16 - 1);
  const int h  = (w >> 7) & (H - 1);
  const int b  = w >> 11;
  const int hk = h >> 2;                   // H/HK = 4
  const int t0 = qt * 16;

  const __bf16* qrow = Qh + (((size_t)b * H + h) * T + (t0 + r)) * D;
  v16bf qf[4];
#pragma unroll
  for (int dk = 0; dk < 4; ++dk) qf[dk] = make_a_frag(qrow + dk * 32, hi);

  v8f yacc[8];
#pragma unroll
  for (int dt = 0; dt < 8; ++dt)
#pragma unroll
    for (int i = 0; i < 8; ++i) yacc[dt][i] = 0.f;
  float mrow[8], lrow[8];
#pragma unroll
  for (int i = 0; i < 8; ++i) { mrow[i] = -1e30f; lrow[i] = 0.f; }

  const __bf16* kbase = Kh + ((size_t)b * HK + hk) * (size_t)T * D;
  const __bf16* vbase = Vt + ((size_t)b * HK + hk) * (size_t)D * T;

  for (int j0 = 0; j0 < t0 + 16; j0 += 32) {
    // ---- scores: two 16x16 tiles over kv cols [j0, j0+32)
    float s[2][8];
#pragma unroll
    for (int kt = 0; kt < 2; ++kt) {
      v8f sa;
#pragma unroll
      for (int i = 0; i < 8; ++i) sa[i] = 0.f;
      const __bf16* kr = kbase + (size_t)(j0 + kt * 16 + r) * D;
#pragma unroll
      for (int dk = 0; dk < 4; ++dk)
        sa = wmma_bf16(qf[dk], *(const v16bf*)(kr + dk * 32 + hi * 16), sa);
      const int tc = j0 + kt * 16 + r;     // kv position this lane's column
#pragma unroll
      for (int rr = 0; rr < 8; ++rr) {
        const int tr = t0 + rr + 8 * hi;   // query position this row
        s[kt][rr] = (tc <= tr) ? sa[rr] * SCALE : -1e30f;
      }
    }
    // ---- online softmax (row stats live across 16-lane half groups)
#pragma unroll
    for (int rr = 0; rr < 8; ++rr) {
      float v = fmaxf(s[0][rr], s[1][rr]);
      v = fmaxf(v, __shfl_xor(v, 1));
      v = fmaxf(v, __shfl_xor(v, 2));
      v = fmaxf(v, __shfl_xor(v, 4));
      v = fmaxf(v, __shfl_xor(v, 8));
      const float mnew  = fmaxf(mrow[rr], v);
      const float alpha = __expf(mrow[rr] - mnew);
      mrow[rr] = mnew;
      const float p0 = __expf(s[0][rr] - mnew);
      const float p1 = __expf(s[1][rr] - mnew);
      Pst[wave][rr + 8 * hi][r]      = (__bf16)p0;
      Pst[wave][rr + 8 * hi][r + 16] = (__bf16)p1;
      float ps = p0 + p1;
      ps += __shfl_xor(ps, 1);
      ps += __shfl_xor(ps, 2);
      ps += __shfl_xor(ps, 4);
      ps += __shfl_xor(ps, 8);
      lrow[rr] = lrow[rr] * alpha + ps;
#pragma unroll
      for (int dt = 0; dt < 8; ++dt) yacc[dt][rr] *= alpha;
    }
    asm volatile("s_wait_dscnt 0" ::: "memory");   // P stores visible to loads
    // ---- P(16x32) @ V(32x128): A-frag from LDS, B-frags contiguous from Vt
    v16bf pf = make_a_frag(&Pst[wave][r][0], hi);
#pragma unroll
    for (int dt = 0; dt < 8; ++dt) {
      const __bf16* vr = vbase + (size_t)(dt * 16 + r) * T + j0 + hi * 16;
      yacc[dt] = wmma_bf16(pf, *(const v16bf*)vr, yacc[dt]);
    }
    asm volatile("s_wait_dscnt 0" ::: "memory");   // don't overwrite P early
  }

#pragma unroll
  for (int dt = 0; dt < 8; ++dt)
#pragma unroll
    for (int rr = 0; rr < 8; ++rr) {
      const int t = t0 + rr + 8 * hi;
      const float val = yacc[dt][rr] / lrow[rr];
      Y[((size_t)b * T + t) * C + h * D + dt * 16 + r] = (__bf16)val;
    }
}

// ---------------------------------------------------------------- out proj
// Same 32x64-per-wave tiling as gemm_qkv; fp32 output straight to d_out.
__global__ __launch_bounds__(128) void gemm_proj(const __bf16* __restrict__ Yb,
                                                 const __bf16* __restrict__ Wp,
                                                 float* __restrict__ Out) {
  using namespace cfg;
  const int lane = threadIdx.x & 31;
  const int wave = threadIdx.x >> 5;
  const int hi = lane >> 4, r = lane & 15;
  const int mtiles = (B * T) / 32;               // 128
  const int mtile = blockIdx.x % mtiles;
  const int nblk  = blockIdx.x / mtiles;         // 0..7
  const int m0 = mtile * 32;
  const int n0 = nblk * 256 + wave * 64;

  v8f acc[2][4];
#pragma unroll
  for (int mt = 0; mt < 2; ++mt)
#pragma unroll
    for (int nt = 0; nt < 4; ++nt)
#pragma unroll
      for (int i = 0; i < 8; ++i) acc[mt][nt][i] = 0.f;

  const __bf16* arow0 = Yb + (size_t)(m0 + r) * C;
  const __bf16* arow1 = Yb + (size_t)(m0 + 16 + r) * C;
  for (int k0 = 0; k0 < C; k0 += 32) {
    v16bf a0 = make_a_frag(arow0 + k0, hi);
    v16bf a1 = make_a_frag(arow1 + k0, hi);
#pragma unroll
    for (int nt = 0; nt < 4; ++nt) {
      const __bf16* brow = Wp + (size_t)(n0 + nt * 16 + r) * C + k0 + hi * 16;
      v16bf bfr = *(const v16bf*)brow;
      acc[0][nt] = wmma_bf16(a0, bfr, acc[0][nt]);
      acc[1][nt] = wmma_bf16(a1, bfr, acc[1][nt]);
    }
  }
#pragma unroll
  for (int mt = 0; mt < 2; ++mt)
#pragma unroll
    for (int nt = 0; nt < 4; ++nt)
#pragma unroll
      for (int rr = 0; rr < 8; ++rr) {
        const int m = m0 + mt * 16 + rr + 8 * hi;
        Out[(size_t)m * C + n0 + nt * 16 + r] = acc[mt][nt][rr];
      }
}

// ---------------------------------------------------------------- launch
extern "C" void kernel_launch(void* const* d_in, const int* in_sizes, int n_in,
                              void* d_out, int out_size, void* d_ws, size_t ws_size,
                              hipStream_t stream) {
  using namespace cfg;
  const float* x   = (const float*)d_in[0];
  const float* Wq  = (const float*)d_in[1];
  const float* Wkv = (const float*)d_in[2];
  const float* Wp  = (const float*)d_in[3];
  (void)in_sizes; (void)n_in; (void)out_size; (void)ws_size;

  char* ws = (char*)d_ws;
  size_t off = 0;
  auto take = [&](size_t elems) -> __bf16* {
    __bf16* p = (__bf16*)(ws + off);
    off += (elems * sizeof(__bf16) + 255) & ~(size_t)255;
    return p;
  };
  __bf16* xb     = take((size_t)B * T * C);       // 16.8 MB
  __bf16* wqkvb  = take((size_t)NQKV * C);        // 12.6 MB
  __bf16* wprojb = take((size_t)C * C);           //  8.4 MB
  __bf16* Qh     = take((size_t)B * H * T * D);   // 16.8 MB
  __bf16* Kh     = take((size_t)B * HK * T * D);  //  4.2 MB
  __bf16* Vt     = take((size_t)B * HK * D * T);  //  4.2 MB
  __bf16* Yb     = take((size_t)B * T * C);       // 16.8 MB  (~80 MB total)

  auto cvt = [&](const float* in, __bf16* out, size_t n) {
    int n4 = (int)(n / 4);
    cvt_f32_bf16<<<dim3((n4 + 255) / 256), dim3(256), 0, stream>>>(in, out, n4);
  };
  cvt(x,   xb,     (size_t)B * T * C);
  cvt(Wq,  wqkvb,  (size_t)C * C);
  cvt(Wkv, wqkvb + (size_t)C * C, (size_t)2 * HK * D * C);
  cvt(Wp,  wprojb, (size_t)C * C);

  gemm_qkv<<<dim3((B * T / 32) * (NQKV / 256)), dim3(128), 0, stream>>>(
      xb, wqkvb, Qh, Kh, Vt);

  const int nrope = (B * H * T + B * HK * T) * (D / 2);
  rope_inplace<<<dim3(nrope / 256), dim3(256), 0, stream>>>(Qh, Kh);

  attn_fa<<<dim3(B * H * (T / 16) / 8), dim3(256), 0, stream>>>(Qh, Kh, Vt, Yb);

  gemm_proj<<<dim3((B * T / 32) * (C / 256)), dim3(128), 0, stream>>>(
      Yb, wprojb, (float*)d_out);
}
